// AAEncoder_69810398429414
// MI455X (gfx1250) — matrix operand order repaired and checked by reading, not compiled, not run on hardware
//
#include <hip/hip_runtime.h>
#include <math.h>

// ---------------- problem constants ----------------
#define TT 20
#define NN 2000
#define TN 40000
#define EE 640000

typedef __attribute__((ext_vector_type(16))) _Float16 v16h;
typedef __attribute__((ext_vector_type(8)))  _Float16 h8;
typedef __attribute__((ext_vector_type(8)))  float    v8f;

union AFrag { v16h v; _Float16 h[16]; h8 q[2]; };
union CFrag { v8f  v; float    f[8];  };

// ---------------- WMMA helpers (wave32, 16x16x32 f16 -> f32) ----------------
// A fragment: 16x32 tile from LDS row-major [64 x 64] halfs (16B-aligned rows).
// Per lane: two contiguous 16B groups -> two ds_load_b128.
__device__ __forceinline__ v16h load_Afrag(const _Float16* s, int rowBase, int kBase) {
  int lane = threadIdx.x & 31;
  const _Float16* p = s + (rowBase + (lane & 15)) * 64 + kBase + ((lane < 16) ? 0 : 16);
  AFrag a;
  a.q[0] = *(const h8*)(p);
  a.q[1] = *(const h8*)(p + 8);
  return a.v;
}

// B fragment from pre-swizzled GLOBAL weights: each 64x64 block stored as
// 8 tiles (tile = kt*4 + nt) x 32 lanes x 16 contiguous halfs.
// Per lane: 32 contiguous bytes -> two global_load_b128, coalesced per wave.
__device__ __forceinline__ v16h load_Bfrag_g(const _Float16* W, int kt, int nt) {
  int lane = threadIdx.x & 31;
  const _Float16* q = W + (((kt << 2) + nt) << 9) + (lane << 4);
  AFrag b;
  b.q[0] = *(const h8*)(q);
  b.q[1] = *(const h8*)(q + 8);
  return b.v;
}

// 64x64 = (64x64 LDS) @ (64x64 swizzled global W) per block (4 waves x 16 rows).
__device__ __forceinline__ void gemm64(const _Float16* A, const _Float16* W,
                                       float* Out, const float* bias, bool accum) {
  int lane    = threadIdx.x & 31;
  int rowBase = (threadIdx.x >> 5) * 16;
  int col0    = lane & 15;
  int r0      = rowBase + ((lane < 16) ? 0 : 8);
#pragma unroll
  for (int nt = 0; nt < 4; ++nt) {
    int col = nt * 16 + col0;
    CFrag c;
    if (accum) {
#pragma unroll
      for (int i = 0; i < 8; ++i) c.f[i] = Out[(r0 + i) * 64 + col];
    } else {
#pragma unroll
      for (int i = 0; i < 8; ++i) c.f[i] = 0.f;
    }
#pragma unroll
    for (int kt = 0; kt < 2; ++kt) {
      v16h a = load_Afrag(A, rowBase, kt * 32);
      v16h b = load_Bfrag_g(W, kt, nt);
      c.v = __builtin_amdgcn_wmma_f32_16x16x32_f16(false, a, false, b, (short)0, c.v, false, false);
    }
    float bv = bias[col];
#pragma unroll
    for (int i = 0; i < 8; ++i) Out[(r0 + i) * 64 + col] = c.f[i] + bv;
  }
}

// Row-wise LayerNorm (+optional ReLU) over 64 cols; in-place f32, optional half copy.
__device__ __forceinline__ void ln_rows(float* Sb, const float* g, const float* bt,
                                        int relu, _Float16* Aout) {
  __syncthreads();
  if (threadIdx.x < 64) {
    float* r = Sb + threadIdx.x * 64;
    float m = 0.f;
#pragma unroll 8
    for (int j = 0; j < 64; ++j) m += r[j];
    m *= (1.f / 64.f);
    float v = 0.f;
#pragma unroll 8
    for (int j = 0; j < 64; ++j) { float d = r[j] - m; v += d * d; }
    v *= (1.f / 64.f);
    float inv = rsqrtf(v + 1e-5f);
#pragma unroll 8
    for (int j = 0; j < 64; ++j) {
      float y = (r[j] - m) * inv * g[j] + bt[j];
      if (relu) y = fmaxf(y, 0.f);
      r[j] = y;
      if (Aout) Aout[threadIdx.x * 64 + j] = (_Float16)y;
    }
  }
  __syncthreads();
}

// LayerNorm reading f32 LDS, writing only a half copy (source preserved).
__device__ __forceinline__ void ln_to_half(const float* Sb, const float* g, const float* bt,
                                           _Float16* Aout) {
  __syncthreads();
  if (threadIdx.x < 64) {
    const float* r = Sb + threadIdx.x * 64;
    float m = 0.f;
#pragma unroll 8
    for (int j = 0; j < 64; ++j) m += r[j];
    m *= (1.f / 64.f);
    float v = 0.f;
#pragma unroll 8
    for (int j = 0; j < 64; ++j) { float d = r[j] - m; v += d * d; }
    v *= (1.f / 64.f);
    float inv = rsqrtf(v + 1e-5f);
#pragma unroll 8
    for (int j = 0; j < 64; ++j)
      Aout[threadIdx.x * 64 + j] = (_Float16)((r[j] - m) * inv * g[j] + bt[j]);
  }
  __syncthreads();
}

__device__ __forceinline__ void atomicMaxF(float* addr, float val) {
  int* ai = (int*)addr;
  float old = *((volatile float*)addr);
  while (val > old) {
    int assumed = __float_as_int(old);
    int prev = atomicCAS(ai, assumed, __float_as_int(val));
    if (prev == assumed) break;
    old = __int_as_float(prev);
  }
}

// ---------------- kernel 0: weight conversion to swizzled f16 fragment layout ----
// Each 64x64 block: out[tile*512 + lane*16 + ii] = Wt[k][n] with
//   kt=tile>>2, nt=tile&3, n = nt*16 + (lane&15), k = kt*32 + (lane<16?0:16) + ii
struct K0Args {
  const float* w[12];       // 12 square [64 out x 64 in] weights
  const float* m1;          // [256 x 64]
  const float* m2;          // [64 x 256]
  _Float16* wt;             // 12 x 4096 swizzled
  _Float16* M1;             // 4 chunks x 4096 swizzled (cols c*64..)
  _Float16* M2;             // 4 chunks x 4096 swizzled (k rows c*64..)
};

__global__ void aae_prep_weights(K0Args a) {
  int i = blockIdx.x * 256 + threadIdx.x;
  if (i < 12 * 4096) {
    int wi = i >> 12, r = i & 4095;
    int tile = r >> 9, lane = (r >> 4) & 31, ii = r & 15;
    int kt = tile >> 2, nt = tile & 3;
    int n = nt * 16 + (lane & 15);
    int k = kt * 32 + ((lane < 16) ? 0 : 16) + ii;
    a.wt[i] = (_Float16)a.w[wi][n * 64 + k];
  } else if (i < 12 * 4096 + 16384) {
    int r = i - 12 * 4096;
    int c = r >> 12, r2 = r & 4095;
    int tile = r2 >> 9, lane = (r2 >> 4) & 31, ii = r2 & 15;
    int kt = tile >> 2, nt = tile & 3;
    int n = c * 64 + nt * 16 + (lane & 15);           // output col of m_w1
    int k = kt * 32 + ((lane < 16) ? 0 : 16) + ii;    // input dim
    a.M1[r] = (_Float16)a.m1[n * 64 + k];
  } else if (i < 12 * 4096 + 32768) {
    int r = i - 12 * 4096 - 16384;
    int c = r >> 12, r2 = r & 4095;
    int tile = r2 >> 9, lane = (r2 >> 4) & 31, ii = r2 & 15;
    int kt = tile >> 2, nt = tile & 3;
    int n = nt * 16 + (lane & 15);                    // output col of m_w2
    int k = c * 64 + kt * 32 + ((lane < 16) ? 0 : 16) + ii;  // hidden dim
    a.M2[r] = (_Float16)a.m2[n * 256 + k];
  }
}

// ---------------- kernel init: clear segment accumulators every call ------------
__global__ void aae_init_accum(float* NUM, float* DEN, float* AMAX) {
  int i = blockIdx.x * 256 + threadIdx.x;
  if (i < TN * 64) NUM[i] = 0.f;
  else if (i < TN * 72) DEN[i - TN * 64] = 0.f;
  else if (i < TN * 80) AMAX[i - TN * 72] = -3.0e38f;
}

// ---------------- kernel 1: node embedding (ce1, ce_n, q) ----------------
struct NodeArgs {
  const float *x, *rot;
  const unsigned char* bos;
  const float *ce_w0, *ce_b0, *ce_g1, *ce_bt1, *ce_b3, *ce_g4, *ce_bt4, *ce_b6, *ce_g7, *ce_bt7;
  const float *bos_tok, *n1_g, *n1_b, *bq;
  const _Float16 *WT_CE3, *WT_CE6, *WT_Q;
  float *CE1, *CEN, *QN;
};

__global__ void __launch_bounds__(128)
aae_node_embed(NodeArgs p) {
  __shared__ __align__(16) _Float16 As[64 * 64];
  __shared__ float S[64 * 64];
  __shared__ float xc2[64][2];

  if (threadIdx.x == 0) {
    __builtin_prefetch(p.WT_CE3, 0, 1);
    __builtin_prefetch(p.WT_CE6, 0, 1);
    __builtin_prefetch(p.WT_Q, 0, 1);
  }

  int base = blockIdx.x * 64;
  if (threadIdx.x < 64) {
    int node = base + threadIdx.x;
    const float* R = p.rot + (node % NN) * 4;
    float x0 = p.x[node * 2], x1 = p.x[node * 2 + 1];
    xc2[threadIdx.x][0] = x0 * R[0] + x1 * R[2];
    xc2[threadIdx.x][1] = x0 * R[1] + x1 * R[3];
  }
  __syncthreads();

  // layer0: 2 -> 64 linear
  for (int i = threadIdx.x; i < 4096; i += 128) {
    int r = i >> 6, j = i & 63;
    S[i] = p.ce_w0[j * 2] * xc2[r][0] + p.ce_w0[j * 2 + 1] * xc2[r][1] + p.ce_b0[j];
  }
  ln_rows(S, p.ce_g1, p.ce_bt1, 1, As);

  gemm64(As, p.WT_CE3, S, p.ce_b3, false);
  ln_rows(S, p.ce_g4, p.ce_bt4, 1, As);

  gemm64(As, p.WT_CE6, S, p.ce_b6, false);
  ln_rows(S, p.ce_g7, p.ce_bt7, 0, nullptr);

  // bos select -> ce1
  for (int i = threadIdx.x; i < 4096; i += 128) {
    int r = i >> 6, j = i & 63;
    int node = base + r;
    int t = node / NN;
    float c = (p.bos[node] != 0) ? p.bos_tok[t * 64 + j] : S[i];
    p.CE1[(size_t)node * 64 + j] = c;
    S[i] = c;
  }
  // norm1 -> ce_n
  ln_rows(S, p.n1_g, p.n1_b, 0, As);
  for (int i = threadIdx.x; i < 4096; i += 128) {
    int r = i >> 6, j = i & 63;
    p.CEN[(size_t)(base + r) * 64 + j] = S[i];
  }
  __syncthreads();
  // q = ce_n @ Wq^T + bq, scaled by 1/sqrt(DH)
  gemm64(As, p.WT_Q, S, p.bq, false);
  __syncthreads();
  for (int i = threadIdx.x; i < 4096; i += 128) {
    int r = i >> 6, j = i & 63;
    p.QN[(size_t)(base + r) * 64 + j] = S[i] * 0.35355339059327373f;
  }
}

// ---------------- kernel 2: edge pipeline (nbr, k, v, alpha, segmax) ------------
struct EdgeArgs {
  const int* ei;
  const float *x, *rot, *eattr;
  const float *nb0_w0, *nb0_b0, *nb0_g, *nb0_bt, *nb0_b1;
  const float *nb1_w0, *nb1_b0, *nb1_g, *nb1_bt, *nb1_b1;
  const float *ag_g0, *ag_b0, *ag_bias, *ag_g1, *ag_b1;
  const float *bk, *bv;
  const _Float16 *WT_NB0, *WT_NB1, *WT_AG, *WT_K, *WT_V;
  const float* QN;
  float *ALPHA, *AMAX;
  _Float16* VE;
};

__global__ void __launch_bounds__(128)
aae_edge(EdgeArgs p) {
  __shared__ __align__(16) _Float16 As[64 * 64];
  __shared__ float S[64 * 64];
  __shared__ float Hsum[64 * 64];
  __shared__ float ab[64][4];
  __shared__ int   srcs[64];

  if (threadIdx.x == 0) {
    __builtin_prefetch(p.WT_NB0, 0, 1);
    __builtin_prefetch(p.WT_NB1, 0, 1);
    __builtin_prefetch(p.WT_AG, 0, 1);
    __builtin_prefetch(p.WT_K, 0, 1);
    __builtin_prefetch(p.WT_V, 0, 1);
  }

  int e0 = blockIdx.x * 64;
  if (threadIdx.x < 64) {
    int e = e0 + threadIdx.x;
    int s = p.ei[e];
    int d = p.ei[EE + e];
    srcs[threadIdx.x] = s;
    const float* R = p.rot + (s % NN) * 4;
    float xj0 = p.x[d * 2], xj1 = p.x[d * 2 + 1];
    ab[threadIdx.x][0] = xj0 * R[0] + xj1 * R[2];
    ab[threadIdx.x][1] = xj0 * R[1] + xj1 * R[3];
    float ea0 = p.eattr[(size_t)e * 2], ea1 = p.eattr[(size_t)e * 2 + 1];
    ab[threadIdx.x][2] = ea0 * R[0] + ea1 * R[2];
    ab[threadIdx.x][3] = ea0 * R[1] + ea1 * R[3];
  }
  __syncthreads();

  // branch 0: lin 2->64, LN, relu, 64x64 GEMM -> Hsum
  for (int i = threadIdx.x; i < 4096; i += 128) {
    int r = i >> 6, j = i & 63;
    S[i] = p.nb0_w0[j * 2] * ab[r][0] + p.nb0_w0[j * 2 + 1] * ab[r][1] + p.nb0_b0[j];
  }
  ln_rows(S, p.nb0_g, p.nb0_bt, 1, As);
  gemm64(As, p.WT_NB0, Hsum, p.nb0_b1, false);

  // branch 1: -> S, then Hsum += S
  __syncthreads();
  for (int i = threadIdx.x; i < 4096; i += 128) {
    int r = i >> 6, j = i & 63;
    S[i] = p.nb1_w0[j * 2] * ab[r][2] + p.nb1_w0[j * 2 + 1] * ab[r][3] + p.nb1_b0[j];
  }
  ln_rows(S, p.nb1_g, p.nb1_bt, 1, As);
  gemm64(As, p.WT_NB1, S, p.nb1_b1, false);
  __syncthreads();
  for (int i = threadIdx.x; i < 4096; i += 128) Hsum[i] += S[i];

  // aggregate: LN, relu, GEMM, LN -> nbr (S f32, As half)
  ln_rows(Hsum, p.ag_g0, p.ag_b0, 1, As);
  gemm64(As, p.WT_AG, S, p.ag_bias, false);
  ln_rows(S, p.ag_g1, p.ag_b1, 0, As);

  // k = nbr @ Wk^T + bk -> Hsum
  gemm64(As, p.WT_K, Hsum, p.bk, false);
  __syncthreads();

  // alpha[e,h] = q[src] . k  (q pre-scaled); atomic segment max
  for (int pp = threadIdx.x; pp < 512; pp += 128) {
    int el = pp >> 3, h = pp & 7;
    int s = srcs[el];
    const float* q = p.QN + (size_t)s * 64 + h * 8;
    const float* k = Hsum + el * 64 + h * 8;
    float dot = 0.f;
#pragma unroll
    for (int dh = 0; dh < 8; ++dh) dot += q[dh] * k[dh];
    p.ALPHA[(size_t)(e0 + el) * 8 + h] = dot;
    atomicMaxF(&p.AMAX[(size_t)s * 8 + h], dot);
  }

  // v = nbr @ Wv^T + bv -> S -> global (half)
  gemm64(As, p.WT_V, S, p.bv, false);
  __syncthreads();
  for (int i = threadIdx.x; i < 4096; i += 128) {
    int r = i >> 6, j = i & 63;
    p.VE[(size_t)(e0 + r) * 64 + j] = (_Float16)S[i];
  }
}

// ---------------- kernel 3: softmax scatter (exp, den, num) ----------------
__global__ void __launch_bounds__(256)
aae_attn_scatter(const int* ei, const float* ALPHA, const float* AMAX,
                 float* DEN, float* NUM, const _Float16* VE) {
  int gid = blockIdx.x * 256 + threadIdx.x;   // gid in [0, EE*8)
  int e = gid >> 3, h = gid & 7;
  int s = ei[e];
  float ex = __expf(ALPHA[gid] - AMAX[(size_t)s * 8 + h]);
  atomicAdd(&DEN[(size_t)s * 8 + h], ex);
  const _Float16* vp = VE + (size_t)e * 64 + h * 8;
  float* np = NUM + (size_t)s * 64 + h * 8;
#pragma unroll
  for (int dh = 0; dh < 8; ++dh) atomicAdd(&np[dh], (float)vp[dh] * ex);
}

// ---------------- kernel 4: gating fusion + output MLP ----------------
struct FinArgs {
  const float *CE1, *CEN, *NUM, *DEN;
  const float *bih, *bhh, *bss, *bo, *n2_g, *n2_b, *m_b1, *m_b2;
  const _Float16 *WT_IH, *WT_HH, *WT_S, *WT_O, *WT_M1, *WT_M2;
  float* out;
};

__global__ void __launch_bounds__(128)
aae_node_final(FinArgs p) {
  __shared__ __align__(16) _Float16 As[64 * 64];
  __shared__ __align__(16) _Float16 A2[64 * 64];
  __shared__ float S[64 * 64];
  __shared__ float OutS[64 * 64];

  if (threadIdx.x == 0) {
    __builtin_prefetch(p.WT_IH, 0, 1);
    __builtin_prefetch(p.WT_HH, 0, 1);
    __builtin_prefetch(p.WT_S, 0, 1);
    __builtin_prefetch(p.WT_M1, 0, 1);
    __builtin_prefetch(p.WT_M2, 0, 1);
  }

  int base = blockIdx.x * 64;
  int lane    = threadIdx.x & 31;
  int rowBase = (threadIdx.x >> 5) * 16;
  int col0    = lane & 15;
  int r0      = rowBase + ((lane < 16) ? 0 : 8);

  // out = num / (den + eps); ce_n half copy
  for (int i = threadIdx.x; i < 4096; i += 128) {
    int r = i >> 6, j = i & 63;
    int node = base + r;
    float o = p.NUM[(size_t)node * 64 + j] / (p.DEN[(size_t)node * 8 + (j >> 3)] + 1e-16f);
    OutS[i] = o;
    As[i] = (_Float16)o;
    A2[i] = (_Float16)p.CEN[(size_t)node * 64 + j];
  }
  __syncthreads();

  // gate = sigmoid(out@Wih + bih + ce_n@Whh + bhh)
  gemm64(As, p.WT_IH, S, p.bih, false);
  gemm64(A2, p.WT_HH, S, p.bhh, true);
  __syncthreads();
  for (int i = threadIdx.x; i < 4096; i += 128) S[i] = 1.f / (1.f + __expf(-S[i]));
  __syncthreads();

  // s = ce_n@Ws + bs; out2 = out + gate*(s - out)  (fused epilogue)
  {
#pragma unroll
    for (int nt = 0; nt < 4; ++nt) {
      CFrag c;
#pragma unroll
      for (int i = 0; i < 8; ++i) c.f[i] = 0.f;
#pragma unroll
      for (int kt = 0; kt < 2; ++kt) {
        v16h a = load_Afrag(A2, rowBase, kt * 32);
        v16h b = load_Bfrag_g(p.WT_S, kt, nt);
        c.v = __builtin_amdgcn_wmma_f32_16x16x32_f16(false, a, false, b, (short)0, c.v, false, false);
      }
      int col = nt * 16 + col0;
#pragma unroll
      for (int i = 0; i < 8; ++i) {
        int idx = (r0 + i) * 64 + col;
        float sv = c.f[i] + p.bss[col];
        float o  = OutS[idx];
        float gt = S[idx];
        OutS[idx] = o + gt * (sv - o);
      }
    }
  }
  __syncthreads();
  for (int i = threadIdx.x; i < 4096; i += 128) As[i] = (_Float16)OutS[i];
  __syncthreads();

  // ce = ce1 + out2@Wo + bo
  gemm64(As, p.WT_O, S, p.bo, false);
  __syncthreads();
  for (int i = threadIdx.x; i < 4096; i += 128) {
    int r = i >> 6, j = i & 63;
    OutS[i] = p.CE1[(size_t)(base + r) * 64 + j] + S[i];
  }

  // ln2 -> half (ce preserved in OutS)
  ln_to_half(OutS, p.n2_g, p.n2_b, A2);

  // MLP 64->256->64 chunked over hidden dim; accumulate in registers
  CFrag acc[4];
#pragma unroll
  for (int nt = 0; nt < 4; ++nt)
#pragma unroll
    for (int i = 0; i < 8; ++i) acc[nt].f[i] = 0.f;

  for (int c = 0; c < 4; ++c) {
    gemm64(A2, p.WT_M1 + c * 4096, S, p.m_b1 + c * 64, false);
    __syncthreads();
    for (int i = threadIdx.x; i < 4096; i += 128) As[i] = (_Float16)fmaxf(S[i], 0.f);
    __syncthreads();
    const _Float16* W2c = p.WT_M2 + c * 4096;
#pragma unroll
    for (int nt = 0; nt < 4; ++nt) {
#pragma unroll
      for (int kt = 0; kt < 2; ++kt) {
        v16h a = load_Afrag(As, rowBase, kt * 32);
        v16h b = load_Bfrag_g(W2c, kt, nt);
        acc[nt].v = __builtin_amdgcn_wmma_f32_16x16x32_f16(false, a, false, b, (short)0, acc[nt].v, false, false);
      }
    }
    __syncthreads();
  }

  // final: out = ce + mlp + m_b2
#pragma unroll
  for (int nt = 0; nt < 4; ++nt) {
    int col = nt * 16 + col0;
#pragma unroll
    for (int i = 0; i < 8; ++i) {
      int row = r0 + i;
      p.out[(size_t)(base + row) * 64 + col] = acc[nt].f[i] + p.m_b2[col] + OutS[row * 64 + col];
    }
  }
}

// ---------------- host launcher ----------------
enum {
  CE_W0 = 0, CE_B0, CE_G1, CE_BT1, CE_W3, CE_B3, CE_G4, CE_BT4, CE_W6, CE_B6, CE_G7, CE_BT7,
  BOS_TOK, N1_G, N1_B,
  NB0_W0, NB0_B0, NB0_G, NB0_BT, NB0_W1, NB0_B1,
  NB1_W0, NB1_B0, NB1_G, NB1_BT, NB1_W1, NB1_B1,
  AG_G0, AG_B0, AG_W, AG_BIAS, AG_G1, AG_B1,
  P_WQ, P_BQ, P_WK, P_BK, P_WV, P_BV, P_WS, P_BS, P_WIH, P_BIH, P_WHH, P_BHH, P_WO, P_BO,
  N2_G, N2_B, M_W1, M_B1, M_W2, M_B2, NPARAMS
};

extern "C" void kernel_launch(void* const* d_in, const int* in_sizes, int n_in,
                              void* d_out, int out_size, void* d_ws, size_t ws_size,
                              hipStream_t stream) {
  (void)in_sizes; (void)n_in; (void)out_size; (void)ws_size;

  const float*          x     = (const float*)d_in[0];
  const int*            ei    = (const int*)d_in[1];
  const float*          eattr = (const float*)d_in[2];
  const unsigned char*  bos   = (const unsigned char*)d_in[3];
  const float*          rot   = (const float*)d_in[4];
  const float* P[NPARAMS];
  for (int i = 0; i < NPARAMS; ++i) P[i] = (const float*)d_in[5 + i];

  // workspace carve-up
  char* w = (char*)d_ws;
  size_t off = 0;
  auto take = [&](size_t b) -> void* {
    void* p = w + off;
    off += (b + 255) & ~(size_t)255;
    return p;
  };
  _Float16* WTsq  = (_Float16*)take((size_t)12 * 4096 * 2);
  _Float16* WT_M1 = (_Float16*)take((size_t)4 * 4096 * 2);
  _Float16* WT_M2 = (_Float16*)take((size_t)4 * 4096 * 2);
  float* CE1   = (float*)take((size_t)TN * 64 * 4);
  float* CEN   = (float*)take((size_t)TN * 64 * 4);
  float* QN    = (float*)take((size_t)TN * 64 * 4);
  float* NUMb  = (float*)take((size_t)TN * 64 * 4);
  float* DENb  = (float*)take((size_t)TN * 8 * 4);
  float* AMAXb = (float*)take((size_t)TN * 8 * 4);
  float* ALPHA = (float*)take((size_t)EE * 8 * 4);
  _Float16* VE = (_Float16*)take((size_t)EE * 64 * 2);

  // weight prep (order: CE3 CE6 NB0 NB1 AG Q K V S IH HH O)
  K0Args a0;
  a0.w[0] = P[CE_W3];  a0.w[1] = P[CE_W6];  a0.w[2] = P[NB0_W1]; a0.w[3] = P[NB1_W1];
  a0.w[4] = P[AG_W];   a0.w[5] = P[P_WQ];   a0.w[6] = P[P_WK];   a0.w[7] = P[P_WV];
  a0.w[8] = P[P_WS];   a0.w[9] = P[P_WIH];  a0.w[10] = P[P_WHH]; a0.w[11] = P[P_WO];
  a0.m1 = P[M_W1]; a0.m2 = P[M_W2];
  a0.wt = WTsq; a0.M1 = WT_M1; a0.M2 = WT_M2;
  aae_prep_weights<<<320, 256, 0, stream>>>(a0);

  aae_init_accum<<<(TN * 80) / 256, 256, 0, stream>>>(NUMb, DENb, AMAXb);

  NodeArgs na;
  na.x = x; na.rot = rot; na.bos = bos;
  na.ce_w0 = P[CE_W0]; na.ce_b0 = P[CE_B0]; na.ce_g1 = P[CE_G1]; na.ce_bt1 = P[CE_BT1];
  na.ce_b3 = P[CE_B3]; na.ce_g4 = P[CE_G4]; na.ce_bt4 = P[CE_BT4];
  na.ce_b6 = P[CE_B6]; na.ce_g7 = P[CE_G7]; na.ce_bt7 = P[CE_BT7];
  na.bos_tok = P[BOS_TOK]; na.n1_g = P[N1_G]; na.n1_b = P[N1_B]; na.bq = P[P_BQ];
  na.WT_CE3 = WTsq + 0 * 4096; na.WT_CE6 = WTsq + 1 * 4096; na.WT_Q = WTsq + 5 * 4096;
  na.CE1 = CE1; na.CEN = CEN; na.QN = QN;
  aae_node_embed<<<TN / 64, 128, 0, stream>>>(na);

  EdgeArgs ea;
  ea.ei = ei; ea.x = x; ea.rot = rot; ea.eattr = eattr;
  ea.nb0_w0 = P[NB0_W0]; ea.nb0_b0 = P[NB0_B0]; ea.nb0_g = P[NB0_G]; ea.nb0_bt = P[NB0_BT]; ea.nb0_b1 = P[NB0_B1];
  ea.nb1_w0 = P[NB1_W0]; ea.nb1_b0 = P[NB1_B0]; ea.nb1_g = P[NB1_G]; ea.nb1_bt = P[NB1_BT]; ea.nb1_b1 = P[NB1_B1];
  ea.ag_g0 = P[AG_G0]; ea.ag_b0 = P[AG_B0]; ea.ag_bias = P[AG_BIAS]; ea.ag_g1 = P[AG_G1]; ea.ag_b1 = P[AG_B1];
  ea.bk = P[P_BK]; ea.bv = P[P_BV];
  ea.WT_NB0 = WTsq + 2 * 4096; ea.WT_NB1 = WTsq + 3 * 4096; ea.WT_AG = WTsq + 4 * 4096;
  ea.WT_K = WTsq + 6 * 4096;   ea.WT_V = WTsq + 7 * 4096;
  ea.QN = QN; ea.ALPHA = ALPHA; ea.AMAX = AMAXb; ea.VE = VE;
  aae_edge<<<EE / 64, 128, 0, stream>>>(ea);

  aae_attn_scatter<<<(EE * 8) / 256, 256, 0, stream>>>(ei, ALPHA, AMAXb, DENb, NUMb, VE);

  FinArgs fa;
  fa.CE1 = CE1; fa.CEN = CEN; fa.NUM = NUMb; fa.DEN = DENb;
  fa.bih = P[P_BIH]; fa.bhh = P[P_BHH]; fa.bss = P[P_BS]; fa.bo = P[P_BO];
  fa.n2_g = P[N2_G]; fa.n2_b = P[N2_B]; fa.m_b1 = P[M_B1]; fa.m_b2 = P[M_B2];
  fa.WT_IH = WTsq + 9 * 4096; fa.WT_HH = WTsq + 10 * 4096;
  fa.WT_S = WTsq + 8 * 4096;  fa.WT_O = WTsq + 11 * 4096;
  fa.WT_M1 = WT_M1; fa.WT_M2 = WT_M2;
  fa.out = (float*)d_out;
  aae_node_final<<<TN / 64, 128, 0, stream>>>(fa);
}
